// MambaBlock_3874060501610
// MI455X (gfx1250) — compile-verified, matrix operand order checked by reading, not run on hardware
//
#include <hip/hip_runtime.h>
#include <hip/hip_fp16.h>

#define H_   1024
#define DI_  2048
#define NS_  16      // state dim N
#define DC_  4
#define DTR_ 64
#define B_   2
#define L_   2048
#define NT_  (B_*L_) // 4096 tokens

typedef _Float16 f16;
typedef __attribute__((ext_vector_type(16))) _Float16 v16h;
typedef __attribute__((ext_vector_type(8)))  float    v8f;

__device__ __forceinline__ float sigmoidf_(float x){ return 1.0f/(1.0f+__expf(-x)); }

#define ACT_NONE     0
#define ACT_SOFTPLUS 1
#define ACT_GELU     2

// Load a 16-element f16 WMMA fragment (A or B) for one lane.
// Layout per ISA 7.12.2 (16-bit A 16x32): lane-half 0 holds K={0..7,16..23},
// half 1 holds K={8..15,24..31} -> two contiguous 16B runs.
__device__ __forceinline__ v16h load_frag(const f16* __restrict__ p, int half) {
  union { v16h v; uint4 q[2]; } u;
  u.q[0] = *(const uint4*)(p + 8*half);
  u.q[1] = *(const uint4*)(p + 16 + 8*half);
  return u.v;
}

// C[M,N] = act(A[M,K] * W[N,K]^T + bias).
// One wave computes a (16*MB) x (16*NB) register tile: A fragments amortized
// over NB WMMAs, B fragments over MB WMMAs -> MB*NB independent accumulator
// chains per k-step (good XDL pipelining, ~21 FLOP/L2-byte at 2x4).
template<int MB, int NB, int ACT, bool HAS_BIAS, bool OUT_F16>
__global__ void gemm_wmma_kernel(const f16* __restrict__ A, int lda,
                                 const f16* __restrict__ W, int ldw,
                                 const float* __restrict__ bias,
                                 float* __restrict__ Cf, f16* __restrict__ Ch,
                                 int ldc, int col_off,
                                 int K, int groups_n, int total_groups)
{
  int wave = threadIdx.x >> 5;
  int grp  = blockIdx.x * (blockDim.x >> 5) + wave;
  if (grp >= total_groups) return;            // wave-uniform: EXEC stays all-1s
  int tmb = (grp / groups_n) * MB;            // first M-tile of this tile
  int tnb = (grp % groups_n) * NB;            // first N-tile of this tile
  int lane = threadIdx.x & 31;
  int half = lane >> 4;
  int r    = lane & 15;
  const f16* Ap[MB];
  #pragma unroll
  for (int j = 0; j < MB; j++)
    Ap[j] = A + (size_t)((tmb + j)*16 + r) * lda;
  const f16* Wp[NB];
  #pragma unroll
  for (int i = 0; i < NB; i++)
    Wp[i] = W + (size_t)((tnb + i)*16 + r) * ldw;

  v8f acc[MB][NB];
  #pragma unroll
  for (int j = 0; j < MB; j++)
    #pragma unroll
    for (int i = 0; i < NB; i++)
      acc[j][i] = (v8f){0.f,0.f,0.f,0.f,0.f,0.f,0.f,0.f};

  for (int k0 = 0; k0 < K; k0 += 32) {
    v16h a[MB];
    #pragma unroll
    for (int j = 0; j < MB; j++) a[j] = load_frag(Ap[j] + k0, half);
    if (k0 + 32 < K) {
      __builtin_prefetch(Ap[0] + k0 + 32, 0, 3);   // near-cache prefetch
      __builtin_prefetch(Wp[0] + k0 + 32, 0, 3);
    }
    #pragma unroll
    for (int i = 0; i < NB; i++) {
      v16h b = load_frag(Wp[i] + k0, half);
      #pragma unroll
      for (int j = 0; j < MB; j++)
        acc[j][i] = __builtin_amdgcn_wmma_f32_16x16x32_f16(
                      false, a[j], false, b, (short)0, acc[j][i], false, false);
    }
  }

  #pragma unroll
  for (int j = 0; j < MB; j++) {
    int mb = (tmb + j)*16 + 8*half;
    #pragma unroll
    for (int i = 0; i < NB; i++) {
      int n = (tnb + i)*16 + r;
      float bv = HAS_BIAS ? bias[n] : 0.f;
      #pragma unroll
      for (int v = 0; v < 8; v++) {
        float c = acc[j][i][v] + bv;
        if (ACT == ACT_SOFTPLUS)  c = (c > 20.f) ? c : log1pf(__expf(c));
        else if (ACT == ACT_GELU) c = 0.5f*c*(1.f + erff(c*0.70710678118f));
        size_t off = (size_t)(mb + v)*ldc + n + col_off;
        if (OUT_F16) Ch[off] = (f16)c; else Cf[off] = c;
      }
    }
  }
}

__global__ void f32_to_f16_kernel(const float* __restrict__ in, f16* __restrict__ out, int n){
  int i = blockIdx.x*blockDim.x + threadIdx.x;
  if (i < n) out[i] = (f16)in[i];
}

// LayerNorm over H=1024, one 256-thread block per token, write f16.
__global__ void ln_kernel(const float* __restrict__ x, const float* __restrict__ g,
                          const float* __restrict__ bta, f16* __restrict__ xn)
{
  int tok = blockIdx.x;
  const float* xr = x + (size_t)tok * H_;
  float v[4], s = 0.f, s2 = 0.f;
  #pragma unroll
  for (int i = 0; i < 4; i++){ v[i] = xr[threadIdx.x + i*256]; s += v[i]; s2 += v[i]*v[i]; }
  __shared__ float sh1[256], sh2[256];
  sh1[threadIdx.x] = s; sh2[threadIdx.x] = s2;
  __syncthreads();
  for (int st = 128; st > 0; st >>= 1){
    if (threadIdx.x < st){ sh1[threadIdx.x] += sh1[threadIdx.x+st];
                           sh2[threadIdx.x] += sh2[threadIdx.x+st]; }
    __syncthreads();
  }
  float mu  = sh1[0] * (1.f/H_);
  float var = sh2[0] * (1.f/H_) - mu*mu;
  float inv = rsqrtf(var + 1e-5f);
  #pragma unroll
  for (int i = 0; i < 4; i++){
    int h = threadIdx.x + i*256;
    xn[(size_t)tok*H_ + h] = (f16)((v[i]-mu)*inv*g[h] + bta[h]);
  }
}

// Depthwise causal conv (DC=4) + SiLU. dir=0 causal, dir=1 anti-causal
// (== flip -> causal conv -> flip, stored token-indexed).
__global__ void conv_silu_kernel(const float* __restrict__ xz, const float* __restrict__ cw,
                                 const float* __restrict__ cb, f16* __restrict__ xi, int dir)
{
  int idx = blockIdx.x*blockDim.x + threadIdx.x;
  if (idx >= B_*L_*DI_) return;
  int d = idx % DI_;
  int t = (idx / DI_) % L_;
  int b = idx / (DI_*L_);
  float acc = cb[d];
  #pragma unroll
  for (int j = 0; j < DC_; j++){
    int tt = dir ? (t + (DC_-1) - j) : (t - (DC_-1) + j);
    if (tt >= 0 && tt < L_)
      acc += cw[d*DC_ + j] * xz[((size_t)b*L_ + tt)*(2*DI_) + d];
  }
  xi[(size_t)idx] = (f16)(acc * sigmoidf_(acc));
}

// Selective scan: one thread per (b, d) channel, 16 states in registers.
// Fuses the y * silu(z) gate; dir=1 traverses time backwards.
__global__ void scan_kernel(const float* __restrict__ delta, const f16* __restrict__ xi,
                            const f16* __restrict__ dbc, const float* __restrict__ A_log,
                            const float* __restrict__ Dv, const float* __restrict__ xz,
                            f16* __restrict__ yg, int dir)
{
  int d = blockIdx.x*blockDim.x + threadIdx.x;
  int b = blockIdx.y;
  float Aa[NS_], hstate[NS_];
  #pragma unroll
  for (int n = 0; n < NS_; n++){ Aa[n] = -__expf(A_log[d*NS_ + n]); hstate[n] = 0.f; }
  float Dd = Dv[d];
  for (int s = 0; s < L_; s++) {
    int t = dir ? (L_-1-s) : s;
    size_t tok = (size_t)b*L_ + t;
    float dl = delta[tok*DI_ + d];
    float u  = (float)xi[tok*DI_ + d];
    const f16* bc = dbc + tok*(DTR_ + 2*NS_) + DTR_;
    float du = dl*u;
    float y  = 0.f;
    #pragma unroll
    for (int n = 0; n < NS_; n++){
      hstate[n] = __expf(dl*Aa[n])*hstate[n] + du*(float)bc[n];
      y = fmaf(hstate[n], (float)bc[NS_ + n], y);
    }
    y = fmaf(u, Dd, y);
    float z = xz[tok*(2*DI_) + DI_ + d];
    yg[tok*DI_ + d] = (f16)(y * (z * sigmoidf_(z)));
  }
}

// out = g[:, :H] * sigmoid(g[:, H:]) + skip
__global__ void glu_gate_kernel(const float* __restrict__ g, const float* __restrict__ skip,
                                float* __restrict__ out)
{
  int i = blockIdx.x*blockDim.x + threadIdx.x;
  if (i >= NT_*H_) return;
  int m = i / H_, h = i % H_;
  float a  = g[(size_t)m*2*H_ + h];
  float gt = g[(size_t)m*2*H_ + H_ + h];
  out[i] = a * sigmoidf_(gt) + skip[i];
}

extern "C" void kernel_launch(void* const* d_in, const int* in_sizes, int n_in,
                              void* d_out, int out_size, void* d_ws, size_t ws_size,
                              hipStream_t stream)
{
  (void)in_sizes; (void)n_in; (void)out_size; (void)ws_size;
  const float* x      = (const float*)d_in[0];
  const float* ln_g   = (const float*)d_in[1];
  const float* ln_b   = (const float*)d_in[2];
  const float* in_w   = (const float*)d_in[3];
  const float* conv_w = (const float*)d_in[4];
  const float* conv_b = (const float*)d_in[5];
  const float* xp_w   = (const float*)d_in[6];
  const float* dt_w   = (const float*)d_in[7];
  const float* dt_b   = (const float*)d_in[8];
  const float* A_log  = (const float*)d_in[9];
  const float* Dv     = (const float*)d_in[10];
  const float* out_w  = (const float*)d_in[11];
  const float* dr_w   = (const float*)d_in[12];
  const float* dr_b   = (const float*)d_in[13];
  const float* glu_w  = (const float*)d_in[14];
  const float* glu_b  = (const float*)d_in[15];
  float* outp = (float*)d_out;

  char* ws = (char*)d_ws;
  size_t off = 0;
  auto take = [&](size_t bytes)->char* {
    char* p = ws + off;
    off += (bytes + 255) & ~(size_t)255;
    return p;
  };
  f16*   xn    = (f16*)  take((size_t)NT_*H_*2);
  f16*   w_in  = (f16*)  take((size_t)2*DI_*H_*2);
  f16*   w_xp  = (f16*)  take((size_t)(DTR_+2*NS_)*DI_*2);
  f16*   w_dt  = (f16*)  take((size_t)DI_*DTR_*2);
  f16*   w_out = (f16*)  take((size_t)H_*DI_*2);
  f16*   w_dr  = (f16*)  take((size_t)H_*2*H_*2);
  f16*   w_glu = (f16*)  take((size_t)2*H_*H_*2);
  float* xz    = (float*)take((size_t)NT_*2*DI_*4);   // also reused for g
  f16*   xi    = (f16*)  take((size_t)NT_*DI_*2);
  f16*   dbc   = (f16*)  take((size_t)NT_*(DTR_+2*NS_)*2);
  float* delta = (float*)take((size_t)NT_*DI_*4);
  f16*   yg    = (f16*)  take((size_t)NT_*DI_*2);
  f16*   xcat  = (f16*)  take((size_t)NT_*2*H_*2);
  f16*   y1    = (f16*)  take((size_t)NT_*H_*2);

  auto cvt = [&](const float* src, f16* dst, int n){
    f32_to_f16_kernel<<<dim3((n+255)/256), dim3(256), 0, stream>>>(src, dst, n);
  };
  cvt(in_w,  w_in,  2*DI_*H_);
  cvt(xp_w,  w_xp,  (DTR_+2*NS_)*DI_);
  cvt(dt_w,  w_dt,  DI_*DTR_);
  cvt(out_w, w_out, H_*DI_);
  cvt(dr_w,  w_dr,  H_*2*H_);
  cvt(glu_w, w_glu, 2*H_*H_);

  ln_kernel<<<dim3(NT_), dim3(256), 0, stream>>>(x, ln_g, ln_b, xn);

  // in_proj: [4096,4096] = xn[4096,1024] @ in_w^T (shared by both directions)
  { int gn = (2*DI_)/16/4, total = (NT_/32)*gn;
    gemm_wmma_kernel<2,4,ACT_NONE,false,false><<<dim3((total+7)/8), dim3(256), 0, stream>>>(
        xn, H_, w_in, H_, nullptr, xz, nullptr, 2*DI_, 0, H_, gn, total); }

  for (int dir = 0; dir < 2; dir++) {
    conv_silu_kernel<<<dim3((B_*L_*DI_+255)/256), dim3(256), 0, stream>>>(
        xz, conv_w, conv_b, xi, dir);
    // x_proj: dbc[4096,96] f16   (N=96 -> 6 tiles -> NB=2)
    { int gn = (DTR_+2*NS_)/16/2, total = (NT_/32)*gn;
      gemm_wmma_kernel<2,2,ACT_NONE,false,true><<<dim3((total+7)/8), dim3(256), 0, stream>>>(
          xi, DI_, w_xp, DI_, nullptr, nullptr, dbc, DTR_+2*NS_, 0, DI_, gn, total); }
    // dt_proj + bias + softplus: delta[4096,2048] f32
    { int gn = DI_/16/4, total = (NT_/32)*gn;
      gemm_wmma_kernel<2,4,ACT_SOFTPLUS,true,false><<<dim3((total+7)/8), dim3(256), 0, stream>>>(
          dbc, DTR_+2*NS_, w_dt, DTR_, dt_b, delta, nullptr, DI_, 0, DTR_, gn, total); }
    scan_kernel<<<dim3(DI_/256, B_), dim3(256), 0, stream>>>(
        delta, xi, dbc, A_log, Dv, xz, yg, dir);
    // out_proj -> concatenated columns of xcat (fwd cols 0..1023, rev 1024..2047)
    { int gn = H_/16/4, total = (NT_/32)*gn;
      gemm_wmma_kernel<2,4,ACT_NONE,false,true><<<dim3((total+7)/8), dim3(256), 0, stream>>>(
          yg, DI_, w_out, DI_, nullptr, nullptr, xcat, 2*H_, dir*H_, DI_, gn, total); }
  }

  // dim_reduce + bias + exact GELU -> y1 f16
  { int gn = H_/16/4, total = (NT_/32)*gn;
    gemm_wmma_kernel<2,4,ACT_GELU,true,true><<<dim3((total+7)/8), dim3(256), 0, stream>>>(
        xcat, 2*H_, w_dr, 2*H_, dr_b, nullptr, y1, H_, 0, 2*H_, gn, total); }
  // glu + bias -> g f32 (aliases xz, which is dead now)
  { int gn = (2*H_)/16/4, total = (NT_/32)*gn;
    gemm_wmma_kernel<2,4,ACT_NONE,true,false><<<dim3((total+7)/8), dim3(256), 0, stream>>>(
        y1, H_, w_glu, H_, glu_b, xz, nullptr, 2*H_, 0, H_, gn, total); }

  glu_gate_kernel<<<dim3((NT_*H_+255)/256), dim3(256), 0, stream>>>(xz, x, outp);
}
